// ParentFinderGRU_28432683499883
// MI455X (gfx1250) — matrix operand order repaired and checked by reading, not compiled, not run on hardware
//
#include <hip/hip_runtime.h>
#include <hip/hip_bf16.h>
#include <math.h>
#include <stdint.h>

#define DEV __device__ __forceinline__

typedef __attribute__((ext_vector_type(16))) __bf16        v16bf;
typedef __attribute__((ext_vector_type(8)))  float         v8f;
typedef __attribute__((ext_vector_type(2)))  float         v2f;
typedef __attribute__((ext_vector_type(4)))  unsigned int  v4u;
typedef __attribute__((ext_vector_type(8)))  unsigned int  v8u;
typedef __attribute__((ext_vector_type(4)))  unsigned int  u32x4;
typedef __attribute__((ext_vector_type(8)))  int           i32x8;
typedef __attribute__((ext_vector_type(4)))  int           i32x4;

// ---- problem dims (fixed by reference) ----
constexpr int B_  = 8;
constexpr int L_  = 2048;
constexpr int H_  = 768;
constexpr int HG  = 512;
constexpr int C_  = 16;
constexpr int S_  = L_ + 1;       // 2049
constexpr int G3  = 3 * HG;       // 1536
constexpr int H2  = H_ / 2;       // 384
constexpr int MX  = B_ * S_;      // 16392
constexpr int ML  = B_ * L_;      // 16384
constexpr float INV_SQRT_HG = 0.044194173824159216f; // 1/sqrt(512)

// ---- helpers ----
DEV unsigned short f2bf(float x) {
  unsigned u = __builtin_bit_cast(unsigned, x);
  unsigned r = u + 0x7FFFu + ((u >> 16) & 1u);
  return (unsigned short)(r >> 16);
}
DEV float sigm(float x) { return 1.0f / (1.0f + __expf(-x)); }

DEV v16bf load_bfrag(const unsigned short* p) {        // 16 contiguous bf16 (32B)
  v8u t = *reinterpret_cast<const v8u*>(p);
  return __builtin_bit_cast(v16bf, t);
}
DEV v16bf load_afrag(const unsigned short* p) {        // K k0..k0+15 (contiguous)
  v4u lo = *reinterpret_cast<const v4u*>(p);
  v4u hi = *reinterpret_cast<const v4u*>(p + 8);
  v8u t;
  t[0]=lo[0]; t[1]=lo[1]; t[2]=lo[2]; t[3]=lo[3];
  t[4]=hi[0]; t[5]=hi[1]; t[6]=hi[2]; t[7]=hi[3];
  return __builtin_bit_cast(v16bf, t);
}
DEV v8f wmma_bf16(v16bf a, v16bf b, v8f c) {
  return __builtin_amdgcn_wmma_f32_16x16x32_bf16(false, a, false, b, (short)0, c, false, false);
}

// ---------------------------------------------------------------------
// Tensor Data Mover: build D# (CDNA5 ISA) and issue
// TENSOR_LOAD_TO_LDS (6-arg toolchain form: g0,g1,g2,g3,extra,cpol).
// data_size = 4B; dims/strides in dword units.
// 2D: tile_d2 == 0.  3D: z uses stride1 (tensor_dim1_stride).
// ---------------------------------------------------------------------
DEV void tdm_load(unsigned lds_byte_off, unsigned long long ga,
                  unsigned tensor_d0, unsigned tensor_d1, unsigned tensor_d2,
                  unsigned tile_d0, unsigned tile_d1, unsigned tile_d2,
                  unsigned long long stride0, unsigned long long stride1) {
  u32x4 g0;
  g0[0] = 1u;                                            // count=1, user mode
  g0[1] = lds_byte_off;                                  // lds_addr
  g0[2] = (unsigned)(ga & 0xFFFFFFFFull);                // global_addr lo
  g0[3] = (unsigned)((ga >> 32) & 0x01FFFFFFull) | 0x80000000u;  // addr hi | type=2
  i32x8 g1;
  g1[0] = 0x20000;                                       // data_size=2 (4B), mask=0
  g1[1] = (int)((tensor_d0 & 0xFFFFu) << 16);            // tensor_dim0 lo16
  g1[2] = (int)(((tensor_d0 >> 16) & 0xFFFFu) | ((tensor_d1 & 0xFFFFu) << 16));
  g1[3] = (int)(((tensor_d1 >> 16) & 0xFFFFu) | ((tile_d0 & 0xFFFFu) << 16));
  g1[4] = (int)((tile_d1 & 0xFFFFu) | ((tile_d2 & 0xFFFFu) << 16));
  g1[5] = (int)(stride0 & 0xFFFFFFFFull);                // tensor_dim0_stride lo32
  g1[6] = (int)(((stride0 >> 32) & 0xFFFFull) | ((stride1 & 0xFFFFull) << 16));
  g1[7] = (int)((stride1 >> 16) & 0xFFFFFFFFull);        // dim1_stride bits 47:16
  i32x4 g2;
  g2[0] = (int)tensor_d2; g2[1] = 0; g2[2] = 0; g2[3] = 0;
  i32x4 g3;
  g3[0] = 0; g3[1] = 0; g3[2] = 0; g3[3] = 0;
  i32x8 gx;                                              // extra words (unused)
  gx[0] = 0; gx[1] = 0; gx[2] = 0; gx[3] = 0;
  gx[4] = 0; gx[5] = 0; gx[6] = 0; gx[7] = 0;
  __builtin_amdgcn_tensor_load_to_lds(g0, g1, g2, g3, gx, 0);
}

// =====================================================================
// init: zero h buffer + barrier counters
// =====================================================================
__global__ void init_k(unsigned int* bar, unsigned short* hbuf) {
  int tid = threadIdx.x;
  if (tid == 0) { bar[0] = 0u; bar[1] = 0u; }
  for (int i = tid; i < 16 * HG; i += 256) hbuf[i] = 0;
}

// =====================================================================
// root masked-mean -> x_bf row 0 of each batch (bf16)
// =====================================================================
__global__ void root_k(const float* __restrict__ lf,
                       const unsigned char* __restrict__ mask,
                       unsigned short* __restrict__ xbf) {
  int b = blockIdx.x, tid = threadIdx.x;
  __shared__ float red[256];
  float cnt = 0.f;
  for (int l = tid; l < L_; l += 256) cnt += mask[(long)b * L_ + l] ? 1.f : 0.f;
  red[tid] = cnt;
  for (int off = 128; off > 0; off >>= 1) {
    __syncthreads();
    if (tid < off) red[tid] += red[tid + off];
  }
  __syncthreads();
  float count = fmaxf(red[0], 1.f);
  for (int h = tid; h < H_; h += 256) {
    float acc = 0.f;
    for (int l = 0; l < L_; ++l)
      if (mask[(long)b * L_ + l]) acc += lf[((long)b * L_ + l) * H_ + h];
    xbf[(long)b * S_ * H_ + h] = f2bf(acc / count);
  }
}

// =====================================================================
// convert line_features f32 -> bf16 into x_with_root rows 1..L
// =====================================================================
__global__ void convx_k(const float* __restrict__ src, unsigned short* __restrict__ xbf) {
  long i = ((long)blockIdx.x * 256 + threadIdx.x) * 4;
  if (i >= (long)B_ * L_ * H_) return;
  int b = (int)(i / ((long)L_ * H_));
  long rem = i - (long)b * L_ * H_;
  const float* s = src + i;
  unsigned short* d = xbf + (long)b * S_ * H_ + H_ + rem;
#pragma unroll
  for (int e = 0; e < 4; ++e) d[e] = f2bf(s[e]);
}

__global__ void conv_k(const float* __restrict__ src, unsigned short* __restrict__ dst, long n) {
  long i = ((long)blockIdx.x * 256 + threadIdx.x) * 4;
  if (i >= n) return;
#pragma unroll
  for (int e = 0; e < 4; ++e) if (i + e < n) dst[i + e] = f2bf(src[i + e]);
}

// =====================================================================
// Generic WMMA-bf16 GEMM:  C[M,N] = act( A[M,K] * W[N,K]^T + bias )
// Block = 256 thr (8 waves), tile 64(M) x 64(N); each wave: 16x32.
// B tile (64 x K bf16) staged in LDS by the Tensor Data Mover; TDM OOB
// semantics zero-fill rows beyond N. A streamed from global + prefetch.
// =====================================================================
template <int OUT_BF16, int RELU>
__global__ __launch_bounds__(256) void gemm_k(
    const unsigned short* __restrict__ A, const unsigned short* __restrict__ Bw,
    const float* __restrict__ bias, float* __restrict__ Cf,
    unsigned short* __restrict__ Cb, int M, int N, int K,
    int abr, long abstride, long abase) {
  extern __shared__ unsigned short lds_b[];       // [64][K] bf16
  const int lane = threadIdx.x & 31;
  const int w = threadIdx.x >> 5;
  const int wm = w & 3, wn = w >> 2;
  const int m0 = blockIdx.y * 64 + wm * 16;
  const int n0blk = blockIdx.x * 64;
  const int l15 = lane & 15, kl = lane >> 4;

  // ---- TDM: stage B tile into LDS (rows beyond N read as zero) ----
  if (w == 0) {
    tdm_load((unsigned)(uintptr_t)lds_b,
             (unsigned long long)(uintptr_t)(Bw + (long)n0blk * K),
             (unsigned)(K >> 1), (unsigned)(N - n0blk), 0u,
             (unsigned)(K >> 1), 64u, 0u,
             (unsigned long long)(K >> 1), 0ull);
    __builtin_amdgcn_s_wait_tensorcnt((short)0);
  }
  __syncthreads();

  int rowA = m0 + l15; if (rowA >= M) rowA = M - 1;
  long aoff = (long)(rowA / abr) * abstride + (long)(rowA % abr) * K + abase;
  const unsigned short* pa = A + aoff;

  const unsigned short* pb0 = lds_b + (long)(wn * 32 + l15) * K;
  const unsigned short* pb1 = lds_b + (long)(wn * 32 + 16 + l15) * K;

  v8f c0 = {}; v8f c1 = {};
  const int KT = K >> 5;
  for (int kt = 0; kt < KT; ++kt) {
    int ka = kt * 32 + kl * 16;
    int kb = kt * 32 + kl * 16;
    v16bf a  = load_afrag(pa + ka);
    v16bf b0 = load_bfrag(pb0 + kb);
    v16bf b1 = load_bfrag(pb1 + kb);
    c0 = wmma_bf16(a, b0, c0);
    c1 = wmma_bf16(a, b1, c1);
    if (kt + 1 < KT) __builtin_prefetch(pa + ka + 32, 0, 1);   // global_prefetch
  }
#pragma unroll
  for (int s = 0; s < 8; ++s) {
    int row = m0 + s + kl * 8;
    if (row >= M) continue;
    int ja = n0blk + wn * 32 + l15, jb = ja + 16;
    if (ja < N) {
      float v = c0[s] + bias[ja];
      if (RELU) v = fmaxf(v, 0.f);
      if (OUT_BF16) Cb[(long)row * N + ja] = f2bf(v);
      else          Cf[(long)row * N + ja] = v;
    }
    if (jb < N) {
      float v = c1[s] + bias[jb];
      if (RELU) v = fmaxf(v, 0.f);
      if (OUT_BF16) Cb[(long)row * N + jb] = f2bf(v);
      else          Cf[(long)row * N + jb] = v;
    }
  }
}

// =====================================================================
// Persistent GRU recurrence: 8 WGs x 128 thr (4 waves). WG g owns h
// columns [g*64, g*64+64); its W_hh slice (3 gates x 64 rows x 512 K,
// bf16 = 196608 B) is DMA'd into LDS by a single 3D TDM descriptor
// (x = 256 dwords/row, y = 64 rows @ row pitch, z = 3 gates @ gate
// pitch). Per step: 48 bf16 WMMAs/wave; h (16x512 bf16, rows 8..15
// zero) exchanged via L2-resident global buffer + global sense barrier.
// =====================================================================
__global__ __launch_bounds__(128) void gru_k(
    const unsigned short* __restrict__ Whh, const float* __restrict__ xg,
    const float* __restrict__ bhh, unsigned short* __restrict__ hbuf,
    unsigned short* __restrict__ gru_bf, unsigned int* __restrict__ bar) {
  extern __shared__ unsigned short ldsw[];          // [3][64][512] bf16
  const int g = blockIdx.x;
  const int tid = threadIdx.x;
  const int lane = tid & 31, w = tid >> 5;
  const int l15 = lane & 15, kl = lane >> 4;

  if (w == 0) {   // one 3D TDM load: rows {j, j+512, j+1024 : j in [g*64, g*64+64)}
    tdm_load((unsigned)(uintptr_t)ldsw,
             (unsigned long long)(uintptr_t)(Whh + (long)g * 64 * HG),
             256u, (unsigned)(512 - g * 64), 3u,
             256u, 64u, 3u,
             256ull, 131072ull /* 512 rows * 256 dwords gate pitch */);
    __builtin_amdgcn_s_wait_tensorcnt((short)0);
  }
  __syncthreads();

  const int colg = g * 64 + w * 16 + l15;                // this lane's h column
  const unsigned short* lbr = ldsw + ((long)(0 * 64 + w * 16 + l15)) * HG;
  const unsigned short* lbz = ldsw + ((long)(1 * 64 + w * 16 + l15)) * HG;
  const unsigned short* lbn = ldsw + ((long)(2 * 64 + w * 16 + l15)) * HG;
  const float bhr = bhh[colg], bhz = bhh[HG + colg], bhn = bhh[2 * HG + colg];

  float hprev[8];
#pragma unroll
  for (int s = 0; s < 8; ++s) hprev[s] = 0.f;

  for (int t = 0; t < S_; ++t) {
    v8f cr = {}, cz = {}, cn = {};
#pragma unroll 4
    for (int kt = 0; kt < (HG >> 5); ++kt) {
      int ka = kt * 32 + kl * 16;
      int kb = kt * 32 + kl * 16;
      v16bf a  = load_afrag(hbuf + (long)l15 * HG + ka);
      cr = wmma_bf16(a, load_bfrag(lbr + kb), cr);
      cz = wmma_bf16(a, load_bfrag(lbz + kb), cz);
      cn = wmma_bf16(a, load_bfrag(lbn + kb), cn);
    }
    if (lane < 16) {                                     // rows 0..7 = batch
#pragma unroll
      for (int s = 0; s < 8; ++s) {
        long base = ((long)s * S_ + t) * G3;
        float xr = xg[base + colg];
        float xz = xg[base + HG + colg];
        float xn = xg[base + 2 * HG + colg];
        float r = sigm(xr + cr[s] + bhr);
        float z = sigm(xz + cz[s] + bhz);
        float n = tanhf(xn + r * (cn[s] + bhn));
        float h = (1.f - z) * n + z * hprev[s];
        hprev[s] = h;
        unsigned short hb = f2bf(h);
        hbuf[(long)s * HG + colg] = hb;
        gru_bf[((long)s * S_ + t) * HG + colg] = hb;
      }
    }
    // ---- 8-WG sense barrier (global gen counter, s_sleep backoff) ----
    __threadfence();
    __syncthreads();
    if (tid == 0) {
      unsigned prev = atomicAdd(&bar[0], 1u);
      if (prev == 7u) {
        atomicExch(&bar[0], 0u);
        __threadfence();
        atomicAdd(&bar[1], 1u);
      } else {
        volatile unsigned* gv = &bar[1];
        while (*gv < (unsigned)(t + 1)) __builtin_amdgcn_s_sleep(1);
      }
    }
    __syncthreads();
    __threadfence();   // acquire other WGs' h stores
  }
}

// =====================================================================
// softmax of cls_logits -> child probs; A2[b,j,:] = probs_with_root @ M_cp
// =====================================================================
__global__ void probs_k(const float* __restrict__ logits,   // [ML,16]
                        const float* __restrict__ Mcp,      // [17,16]
                        float* __restrict__ child, float* __restrict__ A2) {
  int idx = blockIdx.x * blockDim.x + threadIdx.x;
  if (idx >= MX) return;
  int b = idx / S_, s = idx % S_;
  if (s == 0) {
#pragma unroll
    for (int c = 0; c < C_; ++c) {
      child[(long)idx * C_ + c] = 1.f / (float)C_;
      A2[(long)idx * C_ + c] = Mcp[C_ * C_ + c];     // row 16 of M_cp (ROOT class)
    }
    return;
  }
  const float* lg = logits + ((long)b * L_ + (s - 1)) * C_;
  float p[C_], m = -__builtin_inff();
#pragma unroll
  for (int c = 0; c < C_; ++c) m = fmaxf(m, lg[c]);
  float sum = 0.f;
#pragma unroll
  for (int c = 0; c < C_; ++c) { p[c] = __expf(lg[c] - m); sum += p[c]; }
  float inv = 1.f / sum;
#pragma unroll
  for (int c = 0; c < C_; ++c) { p[c] *= inv; child[(long)idx * C_ + c] = p[c]; }
#pragma unroll
  for (int c = 0; c < C_; ++c) {
    float acc = 0.f;
#pragma unroll
    for (int pp = 0; pp < C_; ++pp) acc += p[pp] * Mcp[pp * C_ + c];
    A2[(long)idx * C_ + c] = acc;
  }
}

// =====================================================================
// scores[b,i,j] = q.k^T/sqrt(Hg) + log(clip(child[i,:].A2[j,:])) + masks
// k tile (64 x 512 bf16) staged via TDM; WMMA bf16 K=512 + WMMA f32
// 16x16x4 for the K=16 class-transition term.
// =====================================================================
__global__ __launch_bounds__(256) void scores_k(
    const unsigned short* __restrict__ qm, const unsigned short* __restrict__ km,
    const float* __restrict__ child, const float* __restrict__ A2,
    const unsigned char* __restrict__ mask, float* __restrict__ out) {
  extern __shared__ unsigned short lds_kt[];      // [64][512] bf16
  const int b = blockIdx.z;
  const int lane = threadIdx.x & 31;
  const int w = threadIdx.x >> 5;
  const int wm = w & 3, wn = w >> 2;
  const int i0 = blockIdx.y * 64 + wm * 16;
  const int j0blk = blockIdx.x * 64;
  const int j0 = j0blk + wn * 32;
  const int l15 = lane & 15, kl = lane >> 4;

  if (w == 0) {   // TDM: stage k rows [j0blk, j0blk+64), zero-filled past S
    tdm_load((unsigned)(uintptr_t)lds_kt,
             (unsigned long long)(uintptr_t)(km + ((long)b * S_ + j0blk) * HG),
             256u, (unsigned)(S_ - j0blk), 0u,
             256u, 64u, 0u, 256ull, 0ull);
    __builtin_amdgcn_s_wait_tensorcnt((short)0);
  }
  __syncthreads();

  int ra = i0 + l15;       if (ra >= S_) ra = S_ - 1;
  int rb0 = j0 + l15;      if (rb0 >= S_) rb0 = S_ - 1;   // for A2 epilogue loads
  int rb1 = j0 + 16 + l15; if (rb1 >= S_) rb1 = S_ - 1;
  const unsigned short* pa  = qm + ((long)b * S_ + ra) * HG;
  const unsigned short* pb0 = lds_kt + (long)(wn * 32 + l15) * HG;
  const unsigned short* pb1 = lds_kt + (long)(wn * 32 + 16 + l15) * HG;

  v8f c0 = {}, c1 = {};
#pragma unroll 4
  for (int kt = 0; kt < (HG >> 5); ++kt) {
    int ka = kt * 32 + kl * 16;
    int kb_ = kt * 32 + kl * 16;
    v16bf a = load_afrag(pa + ka);
    c0 = wmma_bf16(a, load_bfrag(pb0 + kb_), c0);
    c1 = wmma_bf16(a, load_bfrag(pb1 + kb_), c1);
  }

  // soft transition term via f32 WMMA 16x16x4 (K = C = 16)
  const float* chr = child + ((long)b * S_ + ra) * C_;
  const float* a20 = A2 + ((long)b * S_ + rb0) * C_;
  const float* a21 = A2 + ((long)b * S_ + rb1) * C_;
  v8f t0 = {}, t1 = {};
#pragma unroll
  for (int kk = 0; kk < 4; ++kk) {
    int k0 = kk * 4 + kl * 2;
    v2f av;  av.x = chr[k0];  av.y = chr[k0 + 1];
    v2f b0v; b0v.x = a20[k0]; b0v.y = a20[k0 + 1];
    v2f b1v; b1v.x = a21[k0]; b1v.y = a21[k0 + 1];
    t0 = __builtin_amdgcn_wmma_f32_16x16x4_f32(false, av, false, b0v, (short)0, t0, false, false);
    t1 = __builtin_amdgcn_wmma_f32_16x16x4_f32(false, av, false, b1v, (short)0, t1, false, false);
  }

#pragma unroll
  for (int s = 0; s < 8; ++s) {
    int i = i0 + s + kl * 8;
    if (i >= S_) continue;
    bool mi = (i == 0) || (mask[(long)b * L_ + (i - 1)] != 0);
    int j = j0 + l15;
    if (j < S_) {
      bool mj = (j == 0) || (mask[(long)b * L_ + (j - 1)] != 0);
      float v = (j > i || !mi || !mj)
                    ? -__builtin_inff()
                    : c0[s] * INV_SQRT_HG + logf(fmaxf(t0[s], 1e-10f));
      out[((long)b * S_ + i) * S_ + j] = v;
    }
    j = j0 + 16 + l15;
    if (j < S_) {
      bool mj = (j == 0) || (mask[(long)b * L_ + (j - 1)] != 0);
      float v = (j > i || !mi || !mj)
                    ? -__builtin_inff()
                    : c1[s] * INV_SQRT_HG + logf(fmaxf(t1[s], 1e-10f));
      out[((long)b * S_ + i) * S_ + j] = v;
    }
  }
}

// =====================================================================
// host launcher
// =====================================================================
extern "C" void kernel_launch(void* const* d_in, const int* in_sizes, int n_in,
                              void* d_out, int out_size, void* d_ws, size_t ws_size,
                              hipStream_t stream) {
  const float* lf   = (const float*)d_in[0];
  const unsigned char* mask = (const unsigned char*)d_in[1];
  const float* W_ih = (const float*)d_in[2];
  const float* W_hh = (const float*)d_in[3];
  const float* b_ih = (const float*)d_in[4];
  const float* b_hh = (const float*)d_in[5];
  const float* Wq   = (const float*)d_in[6];
  const float* bq   = (const float*)d_in[7];
  const float* Wk   = (const float*)d_in[8];
  const float* bk   = (const float*)d_in[9];
  const float* W1   = (const float*)d_in[10];
  const float* b1   = (const float*)d_in[11];
  const float* W2   = (const float*)d_in[12];
  const float* b2   = (const float*)d_in[13];
  const float* Mcp  = (const float*)d_in[14];

  float* out_scores = (float*)d_out;
  float* out_cls    = out_scores + (long)B_ * S_ * S_;

  // ---- workspace layout ----
  char* w = (char*)d_ws;
  size_t off = 0;
  auto alloc = [&](size_t bytes) -> char* {
    char* p = w + off;
    off = (off + bytes + 255) & ~(size_t)255;
    return p;
  };
  unsigned int*   bar     = (unsigned int*)alloc(256);
  unsigned short* x_bf    = (unsigned short*)alloc((size_t)MX * H_ * 2);
  unsigned short* wih_bf  = (unsigned short*)alloc((size_t)G3 * H_ * 2);
  unsigned short* whh_bf  = (unsigned short*)alloc((size_t)G3 * HG * 2);
  unsigned short* wq_bf   = (unsigned short*)alloc((size_t)HG * HG * 2);
  unsigned short* wk_bf   = (unsigned short*)alloc((size_t)HG * HG * 2);
  unsigned short* w1_bf   = (unsigned short*)alloc((size_t)H2 * H_ * 2);
  unsigned short* w2_bf   = (unsigned short*)alloc((size_t)C_ * H2 * 2);
  float*          xg      = (float*)alloc((size_t)MX * G3 * 4);
  unsigned short* hbuf    = (unsigned short*)alloc((size_t)16 * HG * 2);
  unsigned short* gru_bf  = (unsigned short*)alloc((size_t)MX * HG * 2);
  unsigned short* q_bf    = (unsigned short*)alloc((size_t)MX * HG * 2);
  unsigned short* k_bf    = (unsigned short*)alloc((size_t)MX * HG * 2);
  unsigned short* h1_bf   = (unsigned short*)alloc((size_t)ML * H2 * 2);
  float*          child   = (float*)alloc((size_t)MX * C_ * 4);
  float*          A2      = (float*)alloc((size_t)MX * C_ * 4);
  (void)ws_size; (void)in_sizes; (void)n_in; (void)out_size;

  // raise dynamic-LDS caps (TDM-staged tiles)
  (void)hipFuncSetAttribute((const void*)&gemm_k<0, 0>,
                            hipFuncAttributeMaxDynamicSharedMemorySize, 64 * H_ * 2);
  (void)hipFuncSetAttribute((const void*)&gemm_k<1, 1>,
                            hipFuncAttributeMaxDynamicSharedMemorySize, 64 * H_ * 2);
  (void)hipFuncSetAttribute((const void*)&gemm_k<1, 0>,
                            hipFuncAttributeMaxDynamicSharedMemorySize, 64 * HG * 2);
  (void)hipFuncSetAttribute((const void*)&gru_k,
                            hipFuncAttributeMaxDynamicSharedMemorySize, 3 * 64 * HG * 2);
  (void)hipFuncSetAttribute((const void*)&scores_k,
                            hipFuncAttributeMaxDynamicSharedMemorySize, 64 * HG * 2);

  // 1) init barrier + h0
  init_k<<<1, 256, 0, stream>>>(bar, hbuf);
  // 2) root mean + conversions
  root_k<<<B_, 256, 0, stream>>>(lf, mask, x_bf);
  convx_k<<<((long)B_ * L_ * H_ / 4 + 255) / 256, 256, 0, stream>>>(lf, x_bf);
  auto cv = [&](const float* s, unsigned short* d, long n) {
    conv_k<<<(n / 4 + 255) / 256, 256, 0, stream>>>(s, d, n);
  };
  cv(W_ih, wih_bf, (long)G3 * H_);
  cv(W_hh, whh_bf, (long)G3 * HG);
  cv(Wq,   wq_bf,  (long)HG * HG);
  cv(Wk,   wk_bf,  (long)HG * HG);
  cv(W1,   w1_bf,  (long)H2 * H_);
  cv(W2,   w2_bf,  (long)C_ * H2);

  // 3) xg = x_with_root @ W_ih^T + b_ih   [MX,1536] f32
  gemm_k<0, 0><<<dim3(G3 / 64, (MX + 63) / 64), 256, 64 * H_ * 2, stream>>>(
      x_bf, wih_bf, b_ih, xg, nullptr, MX, G3, H_, MX, 0, 0);

  // 4) GRU recurrence (persistent, 8 WGs, W_hh DMA'd into LDS via TDM)
  gru_k<<<8, 128, 3 * 64 * HG * 2, stream>>>(whh_bf, xg, b_hh, hbuf, gru_bf, bar);

  // 5) classification head: h1 = relu(lf @ W1^T + b1) [ML,384] bf16
  gemm_k<1, 1><<<dim3((H2 + 63) / 64, (ML + 63) / 64), 256, 64 * H_ * 2, stream>>>(
      x_bf, w1_bf, b1, nullptr, h1_bf, ML, H2, H_, L_, (long)S_ * H_, (long)H_);
  //    cls_logits = h1 @ W2^T + b2 -> d_out tail  [ML,16] f32
  gemm_k<0, 0><<<dim3(1, (ML + 63) / 64), 256, 64 * H2 * 2, stream>>>(
      h1_bf, w2_bf, b2, out_cls, nullptr, ML, C_, H2, ML, 0, 0);

  // 6) softmax + transition precompute
  probs_k<<<(MX + 255) / 256, 256, 0, stream>>>(out_cls, Mcp, child, A2);

  // 7) q, k projections (bf16 out)
  gemm_k<1, 0><<<dim3(HG / 64, (MX + 63) / 64), 256, 64 * HG * 2, stream>>>(
      gru_bf, wq_bf, bq, nullptr, q_bf, MX, HG, HG, MX, 0, 0);
  gemm_k<1, 0><<<dim3(HG / 64, (MX + 63) / 64), 256, 64 * HG * 2, stream>>>(
      gru_bf, wk_bf, bk, nullptr, k_bf, MX, HG, HG, MX, 0, 0);

  // 8) fused scores + log-transition + masking
  scores_k<<<dim3((S_ + 63) / 64, (S_ + 63) / 64, B_), 256, 64 * HG * 2, stream>>>(
      q_bf, k_bf, child, A2, mask, out_scores);
}
